// Resnet_CRNN_30717606101318
// MI455X (gfx1250) — compile-verified, matrix operand order checked by reading, not run on hardware
//
#include <hip/hip_runtime.h>
#include <hip/hip_bf16.h>

typedef __bf16 bf16;
typedef __attribute__((ext_vector_type(16))) __bf16 v16bf;
typedef __attribute__((ext_vector_type(8)))  __bf16 v8bf;
typedef __attribute__((ext_vector_type(8)))  float   v8f;
typedef int v4i128 __attribute__((vector_size(16)));   // 128-bit payload type for async DMA

#define BM 128
#define BN 128
#define BK 32
#define KPAD 8
#define LDSS (BK + KPAD)   // 40 bf16 per LDS row (80B) -> conflict-free frag reads

// gfx1250 async global->LDS copy path (ASYNCcnt), with sync fallback
#if defined(__has_builtin)
#if __has_builtin(__builtin_amdgcn_global_load_async_to_lds_b128) && \
    __has_builtin(__builtin_amdgcn_s_wait_asynccnt)
#define USE_ASYNC_COPY 1
#endif
#endif
#ifndef USE_ASYNC_COPY
#define USE_ASYNC_COPY 0
#endif

static __device__ inline v8bf zero8() {
  v8bf v;
#pragma unroll
  for (int i = 0; i < 8; ++i) v[i] = (bf16)0.0f;
  return v;
}

static __device__ inline void copy16_g2l(const bf16* g, bf16* l) {
#if USE_ASYNC_COPY
  __builtin_amdgcn_global_load_async_to_lds_b128(
      (__attribute__((address_space(1))) v4i128*)(void*)g,
      (__attribute__((address_space(3))) v4i128*)(void*)l, 0, 0);
#else
  *(v8bf*)l = *(const v8bf*)g;
#endif
}

static __device__ inline void async_fence() {
#if USE_ASYNC_COPY
  __builtin_amdgcn_s_wait_asynccnt(0);
#endif
}

// ---------------------------------------------------------------------------
// Fused WMMA GEMM:  C[M][N] = epilogue( A[M][K] * B[N][K]^T )
//   epilogue: v = acc; v*=scale[n]; v+=bias[n]; v+=resid[m][n]; relu; -> bf16/f32
//   A-row remap (stride>1) implements strided 1x1 shortcut convs in-place.
//   OOB tile rows are CLAMPED (not zeroed): C[r][c] depends only on A[r],B[c],
//   so garbage rows only feed outputs that are never stored.
//   All source addresses are hoisted out of the k-loop: steady state is
//   4 async DMA issues + 12 ds_load_b128 + 8 v_wmma + waits/barrier.
// ---------------------------------------------------------------------------
__global__ __launch_bounds__(256)
void wmma_gemm_bf16_kernel(const bf16* __restrict__ A, const bf16* __restrict__ Bw,
                           const float* __restrict__ scale, const float* __restrict__ bias,
                           const bf16* __restrict__ resid,
                           bf16* __restrict__ outBf, float* __restrict__ outF,
                           int M, int N, int K, int relu,
                           int HoWo, int Wo, int HW, int Wsrc, int stride)
{
  __shared__ bf16 sA[2][BM * LDSS];
  __shared__ bf16 sB[2][BN * LDSS];

  const int tid  = threadIdx.x;
  const int lane = tid & 31;
  const int wave = tid >> 5;
  const int wm   = wave >> 2;   // 0..1  (64 rows each)
  const int wn   = wave & 3;    // 0..3  (32 cols each)
  const long blockM = (long)blockIdx.x * BM;
  const long blockN = (long)blockIdx.y * BN;

  v8f acc[4][2];
#pragma unroll
  for (int i = 0; i < 4; ++i)
#pragma unroll
    for (int j = 0; j < 2; ++j)
#pragma unroll
      for (int r = 0; r < 8; ++r) acc[i][j][r] = 0.0f;

  auto arow = [&](long m) -> long {
    if (stride == 1) return m;
    long b = m / HoWo; long rr = m % HoWo;
    long ho = rr / Wo; long wo = rr % Wo;
    return b * (long)HW + ho * (long)stride * Wsrc + wo * (long)stride;
  };

  // ---- hoisted per-thread tile addressing (k-loop invariant) ----
  const int r0 = tid >> 2;            // (tid*8)/32 : tile row, pass 0
  const int c0 = (tid & 3) * 8;       // (tid*8)%32 : K column within tile
  long gmA0 = blockM + r0;      if (gmA0 >= M) gmA0 = M - 1;
  long gmA1 = blockM + r0 + 64; if (gmA1 >= M) gmA1 = M - 1;
  long gnB0 = blockN + r0;      if (gnB0 >= N) gnB0 = N - 1;
  long gnB1 = blockN + r0 + 64; if (gnB1 >= N) gnB1 = N - 1;
  const bf16* aSrc0 = A  + arow(gmA0) * K + c0;
  const bf16* aSrc1 = A  + arow(gmA1) * K + c0;
  const bf16* bSrc0 = Bw + gnB0 * K + c0;
  const bf16* bSrc1 = Bw + gnB1 * K + c0;
  const int ldst0 = r0 * LDSS + c0;
  const int ldst1 = (r0 + 64) * LDSS + c0;
  long pn = blockN + (lane << 2); if (pn >= N) pn = N - 1;
  const bf16* pfB = Bw + pn * (long)K;   // prefetch base

  auto loadTiles = [&](int buf, int k0) {
    copy16_g2l(aSrc0 + k0, &sA[buf][ldst0]);
    copy16_g2l(aSrc1 + k0, &sA[buf][ldst1]);
    copy16_g2l(bSrc0 + k0, &sB[buf][ldst0]);
    copy16_g2l(bSrc1 + k0, &sB[buf][ldst1]);
  };

  loadTiles(0, 0);
  async_fence();
  __syncthreads();

  const int nk = K / BK;               // all K in this net are multiples of 32
  const int mrow  = lane & 15;
  const int khalf = (lane >> 4) * 16;

  for (int kt = 0; kt < nk; ++kt) {
    const int buf = kt & 1;
    if (kt + 1 < nk) {
      // L2 prefetch hint for upcoming weight tile (emits global_prefetch_b8)
      __builtin_prefetch(pfB + (long)(kt + 1) * BK, 0, 1);
      // fill the other LDS buffer (async DMA overlaps the WMMA chain below)
      loadTiles(buf ^ 1, (kt + 1) * BK);
    }

    v16bf af[4], bfv[2];
#pragma unroll
    for (int i = 0; i < 4; ++i) {
      const bf16* p = &sA[buf][(wm * 64 + i * 16 + mrow) * LDSS + khalf];
      v8bf lo = *(const v8bf*)p;
      v8bf hi = *(const v8bf*)(p + 8);
      af[i] = __builtin_shufflevector(lo, hi, 0,1,2,3,4,5,6,7,8,9,10,11,12,13,14,15);
    }
#pragma unroll
    for (int j = 0; j < 2; ++j) {
      const bf16* p = &sB[buf][(wn * 32 + j * 16 + mrow) * LDSS + khalf];
      v8bf lo = *(const v8bf*)p;
      v8bf hi = *(const v8bf*)(p + 8);
      bfv[j] = __builtin_shufflevector(lo, hi, 0,1,2,3,4,5,6,7,8,9,10,11,12,13,14,15);
    }

#pragma unroll
    for (int i = 0; i < 4; ++i)
#pragma unroll
      for (int j = 0; j < 2; ++j)
        acc[i][j] = __builtin_amdgcn_wmma_f32_16x16x32_bf16(
            false, af[i], false, bfv[j], (short)0, acc[i][j], false, false);

    async_fence();       // next tile's DMA complete before barrier release
    __syncthreads();
  }

  // Epilogue per 16x16 C layout: VGPR r -> M = r + 8*(lane/16), N = lane%16
#pragma unroll
  for (int i = 0; i < 4; ++i) {
#pragma unroll
    for (int j = 0; j < 2; ++j) {
      long col   = blockN + wn * 32 + j * 16 + (lane & 15);
      long rbase = blockM + wm * 64 + i * 16 + ((lane >> 4) << 3);
#pragma unroll
      for (int r = 0; r < 8; ++r) {
        long row = rbase + r;
        if (row < M && col < N) {
          float v = acc[i][j][r];
          if (scale) v *= scale[col];
          if (bias)  v += bias[col];
          if (resid) v += (float)resid[row * N + col];
          if (relu)  v = v > 0.0f ? v : 0.0f;
          if (outBf) outBf[row * N + col] = (bf16)v;
          if (outF)  outF[row * N + col] = v;
        }
      }
    }
  }
}

// ---------------------------------------------------------------------------
// Weight packing: f32 -> bf16 (1x1 & FC are already [N][K]); 3x3 OIHW -> O,tap,I
// ---------------------------------------------------------------------------
__global__ void pack1x1_kernel(const float* __restrict__ w, bf16* __restrict__ o, long n) {
  long t = blockIdx.x * (long)blockDim.x + threadIdx.x;
  if (t < n) o[t] = (bf16)w[t];
}

__global__ void pack3x3_kernel(const float* __restrict__ w, bf16* __restrict__ o, int O, int I) {
  long total = (long)O * I * 9;
  long t = blockIdx.x * (long)blockDim.x + threadIdx.x;
  if (t >= total) return;
  int i   = (int)(t % I);
  long r  = t / I;
  int tap = (int)(r % 9);
  long oo = r / 9;
  o[t] = (bf16)w[(oo * I + i) * 9 + tap];   // src OIHW flat: ((o*I+i)*3+kh)*3+kw
}

// ---------------------------------------------------------------------------
// Stem: 7x7 s2 p3 conv + BN + ReLU.  NCHW f32 in -> NHWC bf16 out.
// ---------------------------------------------------------------------------
__global__ void stem_kernel(const float* __restrict__ x, const float* __restrict__ w,
                            const float* __restrict__ g, const float* __restrict__ b,
                            bf16* __restrict__ out, int B, int H, int W, int Ho, int Wo)
{
  long total = (long)B * Ho * Wo * 64;
  long t = blockIdx.x * (long)blockDim.x + threadIdx.x;
  if (t >= total) return;
  int oc = (int)(t & 63);
  long r = t >> 6;
  int wo = (int)(r % Wo); r /= Wo;
  int ho = (int)(r % Ho);
  int bb = (int)(r / Ho);
  float acc = 0.0f;
  for (int c = 0; c < 3; ++c)
    for (int kh = 0; kh < 7; ++kh) {
      int hy = ho * 2 - 3 + kh;
      if (hy < 0 || hy >= H) continue;
      for (int kw = 0; kw < 7; ++kw) {
        int wx = wo * 2 - 3 + kw;
        if (wx < 0 || wx >= W) continue;
        acc += x[(((long)bb * 3 + c) * H + hy) * W + wx] *
               w[(((long)oc * 3 + c) * 7 + kh) * 7 + kw];
      }
    }
  float v = acc * g[oc] + b[oc];
  out[t] = (bf16)(v > 0.0f ? v : 0.0f);
}

// 3x3 s2 p1 max-pool, NHWC bf16
__global__ void maxpool_kernel(const bf16* __restrict__ x, bf16* __restrict__ out,
                               int B, int H, int W, int C, int Ho, int Wo)
{
  long total = (long)B * Ho * Wo * (C / 8);
  long t = blockIdx.x * (long)blockDim.x + threadIdx.x;
  if (t >= total) return;
  int c8 = (int)(t % (C / 8));
  long r = t / (C / 8);
  int wo = (int)(r % Wo); r /= Wo;
  int ho = (int)(r % Ho);
  int bb = (int)(r / Ho);
  float m[8];
#pragma unroll
  for (int q = 0; q < 8; ++q) m[q] = -3.0e38f;
  for (int kh = 0; kh < 3; ++kh) {
    int hy = ho * 2 - 1 + kh;
    if (hy < 0 || hy >= H) continue;
    for (int kw = 0; kw < 3; ++kw) {
      int wx = wo * 2 - 1 + kw;
      if (wx < 0 || wx >= W) continue;
      v8bf v = *(const v8bf*)(x + (((long)bb * H + hy) * W + wx) * C + c8 * 8);
#pragma unroll
      for (int q = 0; q < 8; ++q) { float f = (float)v[q]; m[q] = f > m[q] ? f : m[q]; }
    }
  }
  v8bf o;
#pragma unroll
  for (int q = 0; q < 8; ++q) o[q] = (bf16)m[q];
  *(v8bf*)(out + (((long)bb * Ho + ho) * Wo + wo) * C + c8 * 8) = o;
}

// im2col for regular 3x3 p1: NHWC bf16 -> cols[M][9*C], tap-major
__global__ void im2col3x3_kernel(const bf16* __restrict__ x, bf16* __restrict__ cols,
                                 int B, int H, int W, int C, int Ho, int Wo, int stride)
{
  long total = (long)B * Ho * Wo * 9 * (C / 8);
  long t = blockIdx.x * (long)blockDim.x + threadIdx.x;
  if (t >= total) return;
  int c8 = (int)(t % (C / 8));
  long r = t / (C / 8);
  int tap = (int)(r % 9);
  long m = r / 9;
  int wo = (int)(m % Wo);
  long r2 = m / Wo;
  int ho = (int)(r2 % Ho);
  int bb = (int)(r2 / Ho);
  int hy = ho * stride - 1 + tap / 3;
  int wx = wo * stride - 1 + tap % 3;
  v8bf v = zero8();
  if (hy >= 0 && hy < H && wx >= 0 && wx < W)
    v = *(const v8bf*)(x + (((long)bb * H + hy) * W + wx) * C + c8 * 8);
  *(v8bf*)(cols + m * (9L * C) + (long)tap * C + c8 * 8) = v;
}

// Tiny 18-channel offset conv (3x3, pad1, stride s): NHWC bf16 in, NCHW f32 out
__global__ void offset_conv_kernel(const bf16* __restrict__ x, const float* __restrict__ w,
                                   const float* __restrict__ bia, float* __restrict__ out,
                                   int B, int H, int W, int C, int Ho, int Wo, int stride)
{
  long total = (long)B * 18 * Ho * Wo;
  long t = blockIdx.x * (long)blockDim.x + threadIdx.x;
  if (t >= total) return;
  int wo = (int)(t % Wo);
  long r = t / Wo;
  int ho = (int)(r % Ho); r /= Ho;
  int oc = (int)(r % 18);
  int bb = (int)(r / 18);
  float acc = bia[oc];
  for (int tap = 0; tap < 9; ++tap) {
    int hy = ho * stride - 1 + tap / 3;
    int wx = wo * stride - 1 + tap % 3;
    if (hy < 0 || hy >= H || wx < 0 || wx >= W) continue;
    const bf16*  xs = x + (((long)bb * H + hy) * W + wx) * C;
    const float* ws = w + (long)oc * C * 9 + tap;      // OIHW: ((oc*C+c)*3+kh)*3+kw
    for (int c = 0; c < C; ++c) acc += (float)xs[c] * ws[(long)c * 9];
  }
  out[t] = acc;
}

// DCNv1 bilinear gather -> cols[M][9*C] (tap-major, same layout as im2col)
__global__ void deform_gather_kernel(const bf16* __restrict__ x, const float* __restrict__ off,
                                     bf16* __restrict__ cols,
                                     int B, int H, int W, int C, int Ho, int Wo, int stride)
{
  long total = (long)B * Ho * Wo * 9 * (C / 8);
  long t = blockIdx.x * (long)blockDim.x + threadIdx.x;
  if (t >= total) return;
  int c8 = (int)(t % (C / 8));
  long r = t / (C / 8);
  int tap = (int)(r % 9);
  long m = r / 9;
  int wo = (int)(m % Wo);
  long r2 = m / Wo;
  int ho = (int)(r2 % Ho);
  int bb = (int)(r2 / Ho);

  float dy = off[(((long)bb * 18 + tap * 2 + 0) * Ho + ho) * Wo + wo];
  float dx = off[(((long)bb * 18 + tap * 2 + 1) * Ho + ho) * Wo + wo];
  float py = (float)(ho * stride - 1 + tap / 3) + dy;
  float px = (float)(wo * stride - 1 + tap % 3) + dx;
  float y0 = floorf(py), x0 = floorf(px);
  float ly = py - y0, lx = px - x0;
  int iy0 = (int)y0, ix0 = (int)x0;

  float acc[8];
#pragma unroll
  for (int q = 0; q < 8; ++q) acc[q] = 0.0f;

  auto corner = [&](int iy, int ix, float wgt) {
    if (iy >= 0 && iy < H && ix >= 0 && ix < W) {
      v8bf v = *(const v8bf*)(x + (((long)bb * H + iy) * W + ix) * C + c8 * 8);
#pragma unroll
      for (int q = 0; q < 8; ++q) acc[q] += wgt * (float)v[q];
    }
  };
  corner(iy0,     ix0,     (1.0f - ly) * (1.0f - lx));
  corner(iy0,     ix0 + 1, (1.0f - ly) * lx);
  corner(iy0 + 1, ix0,     ly * (1.0f - lx));
  corner(iy0 + 1, ix0 + 1, ly * lx);

  v8bf o;
#pragma unroll
  for (int q = 0; q < 8; ++q) o[q] = (bf16)acc[q];
  *(v8bf*)(cols + m * (9L * C) + (long)tap * C + c8 * 8) = o;
}

// log_softmax over rows; GEMM row m = b*Wq + w, output [w][b][n]
__global__ void logsoftmax_kernel(const float* __restrict__ logits, float* __restrict__ out,
                                  int N, int Wq, int Bq)
{
  int m = blockIdx.x;
  const float* src = logits + (long)m * N;
  int w = m % Wq, b = m / Wq;
  float* dst = out + ((long)w * Bq + b) * N;
  __shared__ float red[256];

  float mx = -3.0e38f;
  for (int n = threadIdx.x; n < N; n += 256) mx = fmaxf(mx, src[n]);
  red[threadIdx.x] = mx; __syncthreads();
  for (int s = 128; s > 0; s >>= 1) {
    if (threadIdx.x < s) red[threadIdx.x] = fmaxf(red[threadIdx.x], red[threadIdx.x + s]);
    __syncthreads();
  }
  mx = red[0]; __syncthreads();

  float sum = 0.0f;
  for (int n = threadIdx.x; n < N; n += 256) sum += expf(src[n] - mx);
  red[threadIdx.x] = sum; __syncthreads();
  for (int s = 128; s > 0; s >>= 1) {
    if (threadIdx.x < s) red[threadIdx.x] += red[threadIdx.x + s];
    __syncthreads();
  }
  float lse = logf(red[0]) + mx;
  for (int n = threadIdx.x; n < N; n += 256) dst[n] = src[n] - lse;
}

// ---------------------------------------------------------------------------
// Host orchestration
// ---------------------------------------------------------------------------
static inline unsigned cd(long a, long b) { return (unsigned)((a + b - 1) / b); }

struct Blk {
  const float *b1,*b2,*b3,*db,*dg,*dw,*g1,*g2,*g3,*offb,*offw,*w1,*w2,*w3;
  int planes, cin, stride, dcn, hasdw;
};

extern "C" void kernel_launch(void* const* d_in, const int* in_sizes, int n_in,
                              void* d_out, int out_size, void* d_ws, size_t ws_size,
                              hipStream_t stream) {
  (void)in_sizes; (void)n_in; (void)out_size;
  const float* input = (const float*)d_in[0];
  const float* fc_b  = (const float*)d_in[1];
  const float* fc_w  = (const float*)d_in[2];
  int idx = 3;

  static const int cfg_planes[4] = {64, 128, 256, 512};
  static const int cfg_nb[4]     = {3, 4, 6, 3};
  static const int cfg_stride[4] = {1, 2, 2, 2};

  Blk blks[16]; int nb = 0; int cin = 64;
  for (int li = 0; li < 4; ++li) {
    for (int bi = 0; bi < cfg_nb[li]; ++bi) {
      Blk& B = blks[nb++];
      B.planes = cfg_planes[li];
      B.cin = cin;
      B.stride = (bi == 0) ? cfg_stride[li] : 1;
      B.dcn = (li >= 1);
      B.hasdw = (bi == 0);
      // sorted dict keys: b1,b2,b3,[db,dg,dw],g1,g2,g3,[off_b,off_w],w1,w2,w3
      B.b1 = (const float*)d_in[idx++];
      B.b2 = (const float*)d_in[idx++];
      B.b3 = (const float*)d_in[idx++];
      if (B.hasdw) { B.db = (const float*)d_in[idx++];
                     B.dg = (const float*)d_in[idx++];
                     B.dw = (const float*)d_in[idx++]; }
      else { B.db = B.dg = nullptr; B.dw = nullptr; }
      B.g1 = (const float*)d_in[idx++];
      B.g2 = (const float*)d_in[idx++];
      B.g3 = (const float*)d_in[idx++];
      if (B.dcn) { B.offb = (const float*)d_in[idx++];
                   B.offw = (const float*)d_in[idx++]; }
      else { B.offb = B.offw = nullptr; }
      B.w1 = (const float*)d_in[idx++];
      B.w2 = (const float*)d_in[idx++];
      B.w3 = (const float*)d_in[idx++];
      cin = B.planes * 4;
    }
  }
  const float* stem_b = (const float*)d_in[idx++];
  const float* stem_g = (const float*)d_in[idx++];
  const float* stem_w = (const float*)d_in[idx++];

  // workspace bump allocator
  size_t woff = 0;
  auto alloc = [&](size_t bytes) -> void* {
    void* p = (char*)d_ws + woff;
    woff += (bytes + 255) & ~(size_t)255;
    return p;
  };
  bf16*  act0   = (bf16*)alloc(34ul << 20);
  bf16*  act1   = (bf16*)alloc(34ul << 20);
  bf16*  t1     = (bf16*)alloc(20ul << 20);
  bf16*  sc     = (bf16*)alloc(34ul << 20);
  bf16*  cols   = (bf16*)alloc(80ul << 20);
  bf16*  wpack  = (bf16*)alloc(29ul << 20);
  float* offbuf = (float*)alloc(2ul << 20);
  float* logits = (float*)alloc(28ul << 20);
  if (woff > ws_size) return;   // insufficient scratch: do nothing (deterministic)

  const int B = 64;
  auto g1d = [](long total) { return dim3(cd(total, 256)); };

  // ---- stem + maxpool: 64x3x32x512 -> NHWC bf16 64x8x128x64 ----
  {
    bf16* stemOut = cols;  // temp reuse
    long tot = (long)B * 16 * 256 * 64;
    stem_kernel<<<g1d(tot), 256, 0, stream>>>(input, stem_w, stem_g, stem_b, stemOut,
                                              B, 32, 512, 16, 256);
    long pt = (long)B * 8 * 128 * (64 / 8);
    maxpool_kernel<<<g1d(pt), 256, 0, stream>>>(stemOut, act0, B, 16, 256, 64, 8, 128);
  }

  bf16* act = act0;
  bf16* nxt = act1;
  int H = 8, W = 128;
  cin = 64;

  for (int bi = 0; bi < nb; ++bi) {
    const Blk& K = blks[bi];
    const int p  = K.planes, oc = p * 4, s = K.stride;
    const int Ho = H / s, Wo = W / s;
    const long Min  = (long)B * H * W;
    const long Mout = (long)B * Ho * Wo;

    // ---- w1 1x1 + BN + ReLU ----
    pack1x1_kernel<<<g1d((long)p * cin), 256, 0, stream>>>(K.w1, wpack, (long)p * cin);
    wmma_gemm_bf16_kernel<<<dim3(cd(Min, BM), cd(p, BN)), 256, 0, stream>>>(
        act, wpack, K.g1, K.b1, nullptr, t1, nullptr,
        (int)Min, p, cin, 1, 0, 1, 0, 1, 1);

    // ---- 3x3 (deformable or regular) -> cols ----
    if (K.dcn) {
      long ot = (long)B * 18 * Ho * Wo;
      offset_conv_kernel<<<g1d(ot), 256, 0, stream>>>(t1, K.offw, K.offb, offbuf,
                                                      B, H, W, p, Ho, Wo, s);
      long gt = Mout * 9 * (p / 8);
      deform_gather_kernel<<<g1d(gt), 256, 0, stream>>>(t1, offbuf, cols,
                                                        B, H, W, p, Ho, Wo, s);
    } else {
      long it = Mout * 9 * (p / 8);
      im2col3x3_kernel<<<g1d(it), 256, 0, stream>>>(t1, cols, B, H, W, p, Ho, Wo, s);
    }
    pack3x3_kernel<<<g1d((long)p * p * 9), 256, 0, stream>>>(K.w2, wpack, p, p);
    wmma_gemm_bf16_kernel<<<dim3(cd(Mout, BM), cd(p, BN)), 256, 0, stream>>>(
        cols, wpack, K.g2, K.b2, nullptr, t1, nullptr,
        (int)Mout, p, 9 * p, 1, 0, 1, 0, 1, 1);

    // ---- shortcut ----
    const bf16* res;
    if (K.hasdw) {
      pack1x1_kernel<<<g1d((long)oc * cin), 256, 0, stream>>>(K.dw, wpack, (long)oc * cin);
      wmma_gemm_bf16_kernel<<<dim3(cd(Mout, BM), cd(oc, BN)), 256, 0, stream>>>(
          act, wpack, K.dg, K.db, nullptr, sc, nullptr,
          (int)Mout, oc, cin, 0, Ho * Wo, Wo, H * W, W, s);
      res = sc;
    } else {
      res = act;   // identity: Min == Mout, cin == oc
    }

    // ---- w3 1x1 + BN + residual + ReLU ----
    pack1x1_kernel<<<g1d((long)oc * p), 256, 0, stream>>>(K.w3, wpack, (long)oc * p);
    wmma_gemm_bf16_kernel<<<dim3(cd(Mout, BM), cd(oc, BN)), 256, 0, stream>>>(
        t1, wpack, K.g3, K.b3, res, nxt, nullptr,
        (int)Mout, oc, p, 1, 0, 1, 0, 1, 1);

    { bf16* tmp = act; act = nxt; nxt = tmp; }
    cin = oc; H = Ho; W = Wo;
  }

  // ---- FC: [1024][2048] x [6736][2048]^T + fc_b -> logits f32 ----
  pack1x1_kernel<<<g1d(6736L * 2048), 256, 0, stream>>>(fc_w, wpack, 6736L * 2048);
  wmma_gemm_bf16_kernel<<<dim3(cd(1024, BM), cd(6736, BN)), 256, 0, stream>>>(
      act, wpack, nullptr, fc_b, nullptr, nullptr, logits,
      1024, 6736, 2048, 0, 0, 1, 0, 1, 1);

  // ---- log_softmax, permute rows b*W+w -> [w][b][n] ----
  logsoftmax_kernel<<<dim3(1024), 256, 0, stream>>>(logits, (float*)d_out, 6736, 16, 64);
}
